// PointNet2MSG_55817394979431
// MI455X (gfx1250) — matrix operand order repaired and checked by reading, not compiled, not run on hardware
//
#include <hip/hip_runtime.h>

// ---------------------------------------------------------------------------
// PointNet++ MSG forward for MI455X (gfx1250, wave32).
//
//  * MLP einsums -> v_wmma_f32_16x16x32_f16, 64x32 output block per wave
//    (4 M-tiles x 2 N-tiles = 8 accumulators; A fragments reused across the
//    two B tiles -> ~21 FLOP/byte of per-lane traffic).
//    - activations are f16 with channel dims padded to mult-of-32, zero pad
//      -> A fragment = 2 contiguous global_load_b128 per tile per K-step.
//    - weights pre-packed per layer into the exact B-fragment lane layout
//      -> B fragment = 2 fully coalesced global_load_b128 per K-step.
//    - every GEMM here has M % 64 == 0 -> zero bounds checks in the hot loop.
//  * ball query  -> wave32 ballot-based ordered compaction.
//  * FPS         -> one block per batch, shfl+LDS argmax reduction.
//  Working set ~75MB: fully resident in the 192MB L2.
// ---------------------------------------------------------------------------

typedef __attribute__((ext_vector_type(16))) _Float16 v16h;
typedef __attribute__((ext_vector_type(8)))  _Float16 v8h;
typedef __attribute__((ext_vector_type(8)))  float    v8f;

static __device__ __forceinline__ v16h cat8(v8h lo, v8h hi) {
  return __builtin_shufflevector(lo, hi, 0, 1, 2, 3, 4, 5, 6, 7,
                                 8, 9, 10, 11, 12, 13, 14, 15);
}

// ----------------------------- split pointcloud ----------------------------
__global__ void pc_split_kernel(const float* __restrict__ pc, float* __restrict__ xyz,
                                _Float16* __restrict__ feat, int total) {
  int t = blockIdx.x * blockDim.x + threadIdx.x;
  if (t >= total) return;
  xyz[t * 3 + 0] = pc[t * 4 + 0];
  xyz[t * 3 + 1] = pc[t * 4 + 1];
  xyz[t * 3 + 2] = pc[t * 4 + 2];
  feat[t]        = (_Float16)pc[t * 4 + 3];
}

// ----------------------------- farthest point sampling ---------------------
__global__ void fps_kernel(const float* __restrict__ xyz, int N, int npoint,
                           float* __restrict__ dist, int* __restrict__ out) {
  int b = blockIdx.x;
  const float* X = xyz + (size_t)b * N * 3;
  float* D = dist + (size_t)b * N;
  int* O = out + (size_t)b * npoint;
  int tid = threadIdx.x, nt = blockDim.x;
  __shared__ float swv[32];
  __shared__ int   swi[32];
  __shared__ int   s_last;

  for (int i = tid; i < N; i += nt) D[i] = 1e10f;
  if (tid == 0) { O[0] = 0; s_last = 0; }
  __syncthreads();

  for (int s = 1; s < npoint; ++s) {
    int last = s_last;
    float px = X[last * 3 + 0], py = X[last * 3 + 1], pz = X[last * 3 + 2];
    float bv = -1.0f; int bi = 0;
    for (int i = tid; i < N; i += nt) {
      float dx = X[i * 3 + 0] - px;
      float dy = X[i * 3 + 1] - py;
      float dz = X[i * 3 + 2] - pz;
      float d = dx * dx + dy * dy + dz * dz;
      float dd = D[i];
      if (d < dd) { dd = d; D[i] = d; }
      if (dd > bv || (dd == bv && i < bi)) { bv = dd; bi = i; }
    }
    for (int offd = 16; offd > 0; offd >>= 1) {
      float ov = __shfl_down(bv, offd, 32);
      int   oi = __shfl_down(bi, offd, 32);
      if (ov > bv || (ov == bv && oi < bi)) { bv = ov; bi = oi; }
    }
    int wid = tid >> 5;
    if ((tid & 31) == 0) { swv[wid] = bv; swi[wid] = bi; }
    __syncthreads();
    if (tid < 32) {
      int nw = nt >> 5;
      bv = (tid < nw) ? swv[tid] : -1.0f;
      bi = (tid < nw) ? swi[tid] : 0;
      for (int offd = 16; offd > 0; offd >>= 1) {
        float ov = __shfl_down(bv, offd, 32);
        int   oi = __shfl_down(bi, offd, 32);
        if (ov > bv || (ov == bv && oi < bi)) { bv = ov; bi = oi; }
      }
      if (tid == 0) { O[s] = bi; s_last = bi; }
    }
    __syncthreads();
  }
}

// ----------------------------- gather sampled xyz --------------------------
__global__ void gather_xyz_kernel(const float* __restrict__ src, const int* __restrict__ idx,
                                  int Nsrc, int S, float* __restrict__ dst, int total) {
  int t = blockIdx.x * blockDim.x + threadIdx.x;
  if (t >= total) return;            // total = B*S
  int b = t / S;
  int p = idx[t];
  const float* q = src + ((size_t)b * Nsrc + p) * 3;
  dst[t * 3 + 0] = q[0];
  dst[t * 3 + 1] = q[1];
  dst[t * 3 + 2] = q[2];
}

// ----------------------------- ball query (wave per center) ----------------
__global__ void ball_query_kernel(const float* __restrict__ xyz, const float* __restrict__ nxyz,
                                  int N, int S, int ns, float r2,
                                  int* __restrict__ out, int nCenters) {
  int gw = (int)((blockIdx.x * blockDim.x + threadIdx.x) >> 5);
  if (gw >= nCenters) return;        // wave-uniform
  int lane = threadIdx.x & 31;
  int b = gw / S;
  const float* q = nxyz + (size_t)gw * 3;
  float qx = q[0], qy = q[1], qz = q[2];
  const float* X = xyz + (size_t)b * N * 3;
  int* O = out + (size_t)gw * ns;

  int cnt = 0, first = 0;
  bool haveFirst = false;
  for (int basei = 0; basei < N && cnt < ns; basei += 32) {
    int j = basei + lane;
    bool inr = false;
    if (j < N) {
      float dx = X[j * 3 + 0] - qx;
      float dy = X[j * 3 + 1] - qy;
      float dz = X[j * 3 + 2] - qz;
      inr = (dx * dx + dy * dy + dz * dz) <= r2;
    }
    unsigned m = __builtin_amdgcn_ballot_w32(inr);
    if (!haveFirst && m) { first = basei + __builtin_ctz(m); haveFirst = true; }
    int pos = cnt + __popc(m & ((1u << lane) - 1u));
    if (inr && pos < ns) O[pos] = j;
    cnt += __popc(m);
  }
  int fill0 = cnt < ns ? cnt : ns;
  for (int p = fill0 + lane; p < ns; p += 32) O[p] = first;
}

// ----------------------------- grouping (writes padded f16) ----------------
__global__ void group_kernel(const float* __restrict__ xyz, const _Float16* __restrict__ feats,
                             int C, const float* __restrict__ nxyz, const int* __restrict__ bidx,
                             int N, int S, int ns, int CgP, _Float16* __restrict__ g,
                             long total) {
  long t = (long)blockIdx.x * blockDim.x + threadIdx.x;
  if (t >= total) return;            // total = B*S*ns*CgP
  long bsj = t / CgP;
  int c = (int)(t - bsj * CgP);
  long bs = bsj / ns;
  int j = (int)(bsj - bs * ns);
  int b = (int)(bs / S);
  int p = bidx[bs * ns + j];
  float v = 0.0f;
  if (c < 3)          v = xyz[((size_t)b * N + p) * 3 + c] - nxyz[(size_t)bs * 3 + c];
  else if (c < C + 3) v = (float)feats[((size_t)b * N + p) * C + (c - 3)];
  g[t] = (_Float16)v;
}

// ----------------------------- weight pre-pack -----------------------------
// Packs W[Cin,Cout] (f32) into the per-lane WMMA B-fragment order, f16,
// zero-padded to CinP x CoutP.  Layout: [tile_n][ks][lane][16 halves].
__global__ void wpack_kernel(const float* __restrict__ W, int Cin, int Cout,
                             int CinP, int CoutP, _Float16* __restrict__ Wp, int total) {
  int t = blockIdx.x * blockDim.x + threadIdx.x;
  if (t >= total) return;            // total = CinP*CoutP
  int j = t & 15;
  int lane = (t >> 4) & 31;
  int blk = t >> 9;                  // tn*nks + ks
  int nks = CinP >> 5;
  int tn = blk / nks;
  int ks = blk - tn * nks;
  int col = tn * 16 + (lane & 15);
  int kk = ks * 32 + j + ((lane >> 4) << 4);   // B frag: K = k0 + j + 16*hi
  float v = (kk < Cin && col < Cout) ? W[(size_t)kk * Cout + col] : 0.0f;
  Wp[t] = (_Float16)v;
}

// ----------------------------- WMMA GEMM + bias + relu ---------------------
// Y[M,CoutP](f16) = relu( X[M,CinP](f16) @ Wp + bias ).  M % 64 == 0,
// CoutP % 32 == 0.  One wave computes a 64x32 block (8 D-tiles); the 4 A
// fragments are reused across both B tiles.  Branch-free hot loop.
__global__ void mlp_wmma_kernel(const _Float16* __restrict__ X, const _Float16* __restrict__ Wp,
                                const float* __restrict__ bias, _Float16* __restrict__ Y,
                                int CinP, int Cout, int CoutP, int tiles_m, int tiles_n2) {
  int wave = (int)(((long)blockIdx.x * blockDim.x + threadIdx.x) >> 5);
  if (wave >= tiles_m * tiles_n2) return;    // wave-uniform exit
  int lane = threadIdx.x & 31;
  int hi = lane >> 4;
  int l16 = lane & 15;
  int wm = wave / tiles_n2;
  int tn = wave - wm * tiles_n2;
  int nks = CinP >> 5;

  // A row pointers (rows wm*64 + {0,16,32,48} + l16)
  const v8h* A0 = (const v8h*)X + (((size_t)(wm * 64 + l16) * CinP) >> 3);
  size_t rstep = (size_t)CinP * 2;           // 16 rows in v8h units
  const v8h* Bp0 = (const v8h*)Wp + (((size_t)(2 * tn)     * nks * 32 + lane) << 1);
  const v8h* Bp1 = (const v8h*)Wp + (((size_t)(2 * tn + 1) * nks * 32 + lane) << 1);

  v8f acc00 = {}, acc10 = {}, acc20 = {}, acc30 = {};
  v8f acc01 = {}, acc11 = {}, acc21 = {}, acc31 = {};
  for (int ks = 0; ks < nks; ++ks) {
    int kb = (ks << 2) + (hi << 1);          // lo chunk: halves k0 + 16*hi
    v16h bf0 = cat8(Bp0[0], Bp0[1]);
    v16h bf1 = cat8(Bp1[0], Bp1[1]);
    if (ks + 1 < nks) {
      __builtin_prefetch(Bp0 + 64, 0, 3);
      __builtin_prefetch(Bp1 + 64, 0, 3);
    }
    Bp0 += 64;                               // 32 lanes * 2 v8h per K-step
    Bp1 += 64;
    v16h a0 = cat8(A0[kb],             A0[kb + 1]);
    v16h a1 = cat8(A0[rstep + kb],     A0[rstep + kb + 1]);
    v16h a2 = cat8(A0[2 * rstep + kb], A0[2 * rstep + kb + 1]);
    v16h a3 = cat8(A0[3 * rstep + kb], A0[3 * rstep + kb + 1]);
    acc00 = __builtin_amdgcn_wmma_f32_16x16x32_f16(false, a0, false, bf0, (short)0, acc00, false, false);
    acc01 = __builtin_amdgcn_wmma_f32_16x16x32_f16(false, a0, false, bf1, (short)0, acc01, false, false);
    acc10 = __builtin_amdgcn_wmma_f32_16x16x32_f16(false, a1, false, bf0, (short)0, acc10, false, false);
    acc11 = __builtin_amdgcn_wmma_f32_16x16x32_f16(false, a1, false, bf1, (short)0, acc11, false, false);
    acc20 = __builtin_amdgcn_wmma_f32_16x16x32_f16(false, a2, false, bf0, (short)0, acc20, false, false);
    acc21 = __builtin_amdgcn_wmma_f32_16x16x32_f16(false, a2, false, bf1, (short)0, acc21, false, false);
    acc30 = __builtin_amdgcn_wmma_f32_16x16x32_f16(false, a3, false, bf0, (short)0, acc30, false, false);
    acc31 = __builtin_amdgcn_wmma_f32_16x16x32_f16(false, a3, false, bf1, (short)0, acc31, false, false);
  }

  int col0 = tn * 32 + l16;
  int col1 = col0 + 16;
  float bv0 = (col0 < Cout) ? bias[col0] : 0.0f;   // pad cols -> relu(0+0)=0
  float bv1 = (col1 < Cout) ? bias[col1] : 0.0f;
  int rbase = wm * 64 + (hi << 3);
#pragma unroll
  for (int i = 0; i < 8; ++i) {
    size_t r0 = (size_t)(rbase + i) * CoutP;
    size_t r1 = (size_t)(rbase + 16 + i) * CoutP;
    size_t r2 = (size_t)(rbase + 32 + i) * CoutP;
    size_t r3 = (size_t)(rbase + 48 + i) * CoutP;
    float u;
    u = acc00[i] + bv0; Y[r0 + col0] = (_Float16)(u > 0.0f ? u : 0.0f);
    u = acc01[i] + bv1; Y[r0 + col1] = (_Float16)(u > 0.0f ? u : 0.0f);
    u = acc10[i] + bv0; Y[r1 + col0] = (_Float16)(u > 0.0f ? u : 0.0f);
    u = acc11[i] + bv1; Y[r1 + col1] = (_Float16)(u > 0.0f ? u : 0.0f);
    u = acc20[i] + bv0; Y[r2 + col0] = (_Float16)(u > 0.0f ? u : 0.0f);
    u = acc21[i] + bv1; Y[r2 + col1] = (_Float16)(u > 0.0f ? u : 0.0f);
    u = acc30[i] + bv0; Y[r3 + col0] = (_Float16)(u > 0.0f ? u : 0.0f);
    u = acc31[i] + bv1; Y[r3 + col1] = (_Float16)(u > 0.0f ? u : 0.0f);
  }
}

// ----------------------------- max over samples ----------------------------
__global__ void maxpool_kernel(const _Float16* __restrict__ Y, int ns, int C, int ldY,
                               _Float16* __restrict__ out, int Ctot, int choff, long total) {
  long t = (long)blockIdx.x * blockDim.x + threadIdx.x;
  if (t >= total) return;            // total = B*S*C
  long bs = t / C;
  int c = (int)(t - bs * C);
  float m = -3.0e38f;
  for (int j = 0; j < ns; ++j)
    m = fmaxf(m, (float)Y[((size_t)bs * ns + j) * ldY + c]);
  out[(size_t)bs * Ctot + choff + c] = (_Float16)m;
}

// ----------------------------- 3-NN + inverse-distance weights -------------
__global__ void knn3_kernel(const float* __restrict__ xyz1, const float* __restrict__ xyz2,
                            int N1, int N2, int* __restrict__ kidx, float* __restrict__ kw,
                            int total) {
  int t = blockIdx.x * blockDim.x + threadIdx.x;
  if (t >= total) return;            // total = B*N1
  int b = t / N1;
  const float* p = xyz1 + (size_t)t * 3;
  float px = p[0], py = p[1], pz = p[2];
  const float* q = xyz2 + (size_t)b * N2 * 3;
  float d0 = 3e38f, d1 = 3e38f, d2 = 3e38f;
  int i0 = 0, i1 = 0, i2 = 0;
  for (int j = 0; j < N2; ++j) {
    float dx = px - q[0], dy = py - q[1], dz = pz - q[2];
    q += 3;
    float d = dx * dx + dy * dy + dz * dz;
    if (d < d0)      { d2 = d1; i2 = i1; d1 = d0; i1 = i0; d0 = d; i0 = j; }
    else if (d < d1) { d2 = d1; i2 = i1; d1 = d;  i1 = j; }
    else if (d < d2) { d2 = d;  i2 = j; }
  }
  float w0 = 1.0f / (d0 + 1e-8f);
  float w1 = 1.0f / (d1 + 1e-8f);
  float w2 = 1.0f / (d2 + 1e-8f);
  float s = w0 + w1 + w2;
  kidx[t * 3 + 0] = i0; kidx[t * 3 + 1] = i1; kidx[t * 3 + 2] = i2;
  kw[t * 3 + 0] = w0 / s; kw[t * 3 + 1] = w1 / s; kw[t * 3 + 2] = w2 / s;
}

// ----------------------------- interpolate + concat (padded f16) -----------
__global__ void interp_concat_kernel(const _Float16* __restrict__ f2, int C2,
                                     const _Float16* __restrict__ f1, int C1,
                                     const int* __restrict__ kidx, const float* __restrict__ kw,
                                     int N1, int N2, int CtP, _Float16* __restrict__ out,
                                     long total) {
  long t = (long)blockIdx.x * blockDim.x + threadIdx.x;
  if (t >= total) return;            // total = B*N1*CtP
  long bn = t / CtP;
  int c = (int)(t - bn * CtP);
  int b = (int)(bn / N1);
  float v = 0.0f;
  if (c < C2) {
    const _Float16* base2 = f2 + (size_t)b * N2 * C2;
    int i0 = kidx[bn * 3 + 0], i1 = kidx[bn * 3 + 1], i2 = kidx[bn * 3 + 2];
    float w0 = kw[bn * 3 + 0], w1 = kw[bn * 3 + 1], w2 = kw[bn * 3 + 2];
    v = w0 * (float)base2[(size_t)i0 * C2 + c] + w1 * (float)base2[(size_t)i1 * C2 + c] +
        w2 * (float)base2[(size_t)i2 * C2 + c];
  } else if (c < C2 + C1) {
    v = (float)f1[bn * C1 + (c - C2)];
  }
  out[t] = (_Float16)v;
}

// ----------------------------- final max over N ----------------------------
__global__ void global_max_kernel(const _Float16* __restrict__ f, int Npts, int C,
                                  float* __restrict__ out) {
  int b = blockIdx.x / C, c = blockIdx.x % C;
  __shared__ float sm[256];
  float m = -3.0e38f;
  for (int n = threadIdx.x; n < Npts; n += blockDim.x)
    m = fmaxf(m, (float)f[((size_t)b * Npts + n) * C + c]);
  sm[threadIdx.x] = m;
  __syncthreads();
  for (int s = 128; s > 0; s >>= 1) {
    if ((int)threadIdx.x < s) sm[threadIdx.x] = fmaxf(sm[threadIdx.x], sm[threadIdx.x + s]);
    __syncthreads();
  }
  if (threadIdx.x == 0) out[blockIdx.x] = sm[0];
}

// ===========================================================================
extern "C" void kernel_launch(void* const* d_in, const int* in_sizes, int n_in,
                              void* d_out, int out_size, void* d_ws, size_t ws_size,
                              hipStream_t stream) {
  (void)in_sizes; (void)n_in; (void)out_size; (void)ws_size;
  const int B = 2, N0 = 16384;
  const int Np[5] = {16384, 4096, 1024, 256, 64};
  const int Cf[5] = {1, 96, 256, 512, 1024};
  const float radii[4][2] = {{0.1f, 0.5f}, {0.5f, 1.0f}, {1.0f, 2.0f}, {2.0f, 4.0f}};
  const int nsamp[2] = {16, 32};
  const int sa_ch[4][2][4] = {
      {{4, 16, 16, 32}, {4, 32, 32, 64}},
      {{99, 64, 64, 128}, {99, 64, 96, 128}},
      {{259, 128, 196, 256}, {259, 128, 196, 256}},
      {{515, 256, 256, 512}, {515, 256, 384, 512}}};
  const int fp_ch[4][3] = {{257, 128, 128}, {608, 256, 256}, {768, 512, 512}, {1536, 512, 512}};
  auto pad32 = [](int c) { return (c + 31) & ~31; };

  const float* pc = (const float*)d_in[0];
  auto saW = [&](int l, int br, int ly) { return (const float*)d_in[1 + (l * 2 + br) * 6 + ly * 2]; };
  auto saB = [&](int l, int br, int ly) { return (const float*)d_in[2 + (l * 2 + br) * 6 + ly * 2]; };
  auto fpW = [&](int i, int ly) { return (const float*)d_in[49 + i * 4 + ly * 2]; };
  auto fpB = [&](int i, int ly) { return (const float*)d_in[50 + i * 4 + ly * 2]; };

  // workspace bump allocator
  char* basep = (char*)d_ws;
  size_t off = 0;
  auto carveF = [&](size_t n) -> float* {
    float* p = (float*)(basep + off);
    off = (off + n * sizeof(float) + 255) & ~(size_t)255;
    return p;
  };
  auto carveI = [&](size_t n) -> int* {
    int* p = (int*)(basep + off);
    off = (off + n * sizeof(int) + 255) & ~(size_t)255;
    return p;
  };
  auto carveH = [&](size_t n) -> _Float16* {
    _Float16* p = (_Float16*)(basep + off);
    off = (off + n * sizeof(_Float16) + 255) & ~(size_t)255;
    return p;
  };

  float* xyzL[5];
  _Float16* featL[5];
  xyzL[0]  = carveF((size_t)B * N0 * 3);
  featL[0] = carveH((size_t)B * N0);
  for (int l = 1; l < 5; ++l) xyzL[l]  = carveF((size_t)B * Np[l] * 3);
  for (int l = 1; l < 5; ++l) featL[l] = carveH((size_t)B * Np[l] * Cf[l]);
  float*    fpsDist = carveF((size_t)B * N0);
  int*      fpsIdx  = carveI((size_t)B * 4096);
  int*      ballIdx = carveI((size_t)B * 4096 * 32);
  int*      knnIdx  = carveI((size_t)B * N0 * 3);
  float*    knnW    = carveF((size_t)B * N0 * 3);
  _Float16* wpack   = carveH((size_t)1536 * 512);       // max CinP*CoutP
  _Float16* bufA    = carveH((size_t)10485760);         // peak 32768x288 (fp0 concat)
  _Float16* bufB    = carveH((size_t)16842752);         // peak 262144x64 (SA L0 b1 l2)
  _Float16* fpf[4];
  fpf[0] = carveH((size_t)B * 16384 * 128);
  fpf[1] = carveH((size_t)B * 4096 * 256);
  fpf[2] = carveH((size_t)B * 1024 * 512);
  fpf[3] = carveH((size_t)B * 256 * 512);

  auto blocks1d = [](long total, int bs) { return (unsigned)((total + bs - 1) / bs); };

  auto run_mlp = [&](const _Float16* X, const float* Wsrc, const float* bias, _Float16* Y,
                     long M, int Cin, int Cout) {
    int CinP = pad32(Cin), CoutP = pad32(Cout);
    int wtot = CinP * CoutP;
    wpack_kernel<<<blocks1d(wtot, 256), 256, 0, stream>>>(Wsrc, Cin, Cout, CinP, CoutP,
                                                          wpack, wtot);
    int tiles_m = (int)(M / 64);      // M % 64 == 0 for every layer in this net
    int tiles_n2 = CoutP / 32;        // CoutP % 32 == 0 by construction
    long waves = (long)tiles_m * tiles_n2;
    mlp_wmma_kernel<<<blocks1d(waves, 8), 256, 0, stream>>>(X, wpack, bias, Y, CinP, Cout,
                                                            CoutP, tiles_m, tiles_n2);
  };

  pc_split_kernel<<<blocks1d((long)B * N0, 256), 256, 0, stream>>>(pc, xyzL[0], featL[0], B * N0);

  // ---------------- set abstraction levels ----------------
  for (int l = 0; l < 4; ++l) {
    int N = Np[l], S = Np[l + 1], C = Cf[l], Ctot = Cf[l + 1];
    fps_kernel<<<B, 1024, 0, stream>>>(xyzL[l], N, S, fpsDist, fpsIdx);
    gather_xyz_kernel<<<blocks1d((long)B * S, 256), 256, 0, stream>>>(xyzL[l], fpsIdx, N, S,
                                                                      xyzL[l + 1], B * S);
    int choff = 0;
    for (int br = 0; br < 2; ++br) {
      int ns = nsamp[br];
      float r = radii[l][br];
      int nC = B * S;
      ball_query_kernel<<<blocks1d((long)nC * 32, 256), 256, 0, stream>>>(
          xyzL[l], xyzL[l + 1], N, S, ns, r * r, ballIdx, nC);
      const int* ch = sa_ch[l][br];
      int CgP = pad32(ch[0]);         // == pad32(C+3)
      long gTot = (long)B * S * ns * CgP;
      group_kernel<<<blocks1d(gTot, 256), 256, 0, stream>>>(xyzL[l], featL[l], C, xyzL[l + 1],
                                                            ballIdx, N, S, ns, CgP, bufA, gTot);
      long M = (long)B * S * ns;
      run_mlp(bufA, saW(l, br, 0), saB(l, br, 0), bufB, M, ch[0], ch[1]);
      run_mlp(bufB, saW(l, br, 1), saB(l, br, 1), bufA, M, ch[1], ch[2]);
      run_mlp(bufA, saW(l, br, 2), saB(l, br, 2), bufB, M, ch[2], ch[3]);
      long pTot = (long)B * S * ch[3];
      maxpool_kernel<<<blocks1d(pTot, 256), 256, 0, stream>>>(bufB, ns, ch[3], pad32(ch[3]),
                                                              featL[l + 1], Ctot, choff, pTot);
      choff += ch[3];
    }
  }

  // ---------------- feature propagation levels ----------------
  for (int i = 3; i >= 0; --i) {
    int N1 = Np[i], N2 = Np[i + 1];
    const _Float16* f2 = (i == 3) ? featL[4] : fpf[i + 1];
    int C2 = (i == 3) ? Cf[4] : fp_ch[i + 1][2];   // 1024 / 512 / 512 / 256, all %32==0
    const _Float16* f1 = featL[i];
    int C1 = Cf[i];
    knn3_kernel<<<blocks1d((long)B * N1, 256), 256, 0, stream>>>(xyzL[i], xyzL[i + 1], N1, N2,
                                                                 knnIdx, knnW, B * N1);
    int CtP = pad32(C2 + C1);         // == pad32(fp_ch[i][0])
    long iTot = (long)B * N1 * CtP;
    interp_concat_kernel<<<blocks1d(iTot, 256), 256, 0, stream>>>(f2, C2, f1, C1, knnIdx, knnW,
                                                                  N1, N2, CtP, bufA, iTot);
    long M = (long)B * N1;
    run_mlp(bufA, fpW(i, 0), fpB(i, 0), bufB, M, fp_ch[i][0], fp_ch[i][1]);
    run_mlp(bufB, fpW(i, 1), fpB(i, 1), fpf[i], M, fp_ch[i][1], fp_ch[i][2]);
  }

  // ---------------- global max over points ----------------
  global_max_kernel<<<B * 128, 256, 0, stream>>>(fpf[0], N0, 128, (float*)d_out);
}
